// MemorizingGPT_87316685128020
// MI455X (gfx1250) — matrix operand order repaired and checked by reading, not compiled, not run on hardware
//
#include <hip/hip_runtime.h>

// ---------- types ----------
typedef __bf16 v16bf __attribute__((ext_vector_type(16)));
typedef __bf16 v8bf  __attribute__((ext_vector_type(8)));
typedef float  v8f   __attribute__((ext_vector_type(8)));

#define T_LEN 2048
#define D_DIM 2048
#define H_NUM 16
#define HD_DIM 128

__device__ __forceinline__ unsigned short f32_to_bf16(float f) {
    unsigned int u = __float_as_uint(f);
    unsigned int r = u + 0x7FFFu + ((u >> 16) & 1u);   // round-to-nearest-even
    return (unsigned short)(r >> 16);
}
__device__ __forceinline__ unsigned int pack2_bf16(float a, float b) {
    return (unsigned int)f32_to_bf16(a) | ((unsigned int)f32_to_bf16(b) << 16);
}

// ---------- block-wide sum reduce (256 thr) ----------
__device__ __forceinline__ float block_reduce_sum(float v, float* smem, int tid) {
    #pragma unroll
    for (int m = 16; m >= 1; m >>= 1) v += __shfl_xor(v, m, 32);
    int wid = tid >> 5;
    if ((tid & 31) == 0) smem[wid] = v;
    __syncthreads();
    float s = 0.f;
    if (tid < 8) {
        s = smem[tid];
        #pragma unroll
        for (int m = 4; m >= 1; m >>= 1) s += __shfl_xor(s, m, 32);
        if (tid == 0) smem[0] = s;
    }
    __syncthreads();
    float r = smem[0];
    __syncthreads();
    return r;
}

// ---------- fused residual blend + rms_norm (row per block) ----------
__global__ __launch_bounds__(256) void blend_rmsnorm_kernel(
    const float* __restrict__ x, const float* __restrict__ x0,
    const float* __restrict__ lambdas, float* __restrict__ out) {
    __shared__ float red[8];
    int t = blockIdx.x, tid = threadIdx.x;
    float l0 = lambdas[0], l1 = lambdas[1];
    const float* xr  = x  + (size_t)t * D_DIM;
    const float* x0r = x0 + (size_t)t * D_DIM;
    float vals[8]; float ss = 0.f;
    #pragma unroll
    for (int j = 0; j < 8; j++) {
        int i = tid + j * 256;
        float v = l0 * xr[i] + l1 * x0r[i];
        vals[j] = v; ss += v * v;
    }
    float tot = block_reduce_sum(ss, red, tid);
    float rms = rsqrtf(tot * (1.f / (float)D_DIM) + 1.1920929e-7f);
    float* o = out + (size_t)t * D_DIM;
    #pragma unroll
    for (int j = 0; j < 8; j++) o[tid + j * 256] = vals[j] * rms;
}

__global__ __launch_bounds__(256) void rmsnorm_kernel(
    const float* __restrict__ src, float* __restrict__ out) {
    __shared__ float red[8];
    int t = blockIdx.x, tid = threadIdx.x;
    const float* xr = src + (size_t)t * D_DIM;
    float vals[8]; float ss = 0.f;
    #pragma unroll
    for (int j = 0; j < 8; j++) {
        float v = xr[tid + j * 256];
        vals[j] = v; ss += v * v;
    }
    float tot = block_reduce_sum(ss, red, tid);
    float rms = rsqrtf(tot * (1.f / (float)D_DIM) + 1.1920929e-7f);
    float* o = out + (size_t)t * D_DIM;
    #pragma unroll
    for (int j = 0; j < 8; j++) o[tid + j * 256] = vals[j] * rms;
}

// ---------- per-head rms_norm + rotary; f32 [T][D] -> bf16 [H][T][HD] ----------
__global__ __launch_bounds__(256) void rope_norm_kernel(
    const float* __restrict__ src, unsigned short* __restrict__ dst) {
    int t = blockIdx.x, tid = threadIdx.x;
    int h = tid >> 4, ln = tid & 15;            // 16 lanes per head
    const float* row = src + (size_t)t * D_DIM + h * HD_DIM;
    float ss = 0.f;
    #pragma unroll
    for (int j = 0; j < 8; j++) {
        float v = row[ln + j * 16];
        ss += v * v;
    }
    #pragma unroll
    for (int m = 8; m >= 1; m >>= 1) ss += __shfl_xor(ss, m, 32); // 16-lane half = one head
    float rms = rsqrtf(ss * (1.f / (float)HD_DIM) + 1.1920929e-7f);
    float tt = (float)t;
    const float kfreq = -0.14391156831212787f;  // -ln(10000)/64
    #pragma unroll
    for (int j = 0; j < 8; j++) {
        int hd = ln + j * 16;
        int p  = hd & 63;
        float x1 = row[p]      * rms;
        float x2 = row[p + 64] * rms;
        float fr = tt * __expf((float)p * kfreq);
        float c = __cosf(fr), s = __sinf(fr);
        float o = (hd < 64) ? (x1 * c + x2 * s) : (-x1 * s + x2 * c);
        dst[((size_t)h * T_LEN + t) * HD_DIM + hd] = f32_to_bf16(o);
    }
}

// ---------- value blend: f32 [T][D] + vi -> bf16 [H][T][HD] ----------
__global__ __launch_bounds__(256) void vblend_kernel(
    const float* __restrict__ v, const float* __restrict__ vi,
    const float* __restrict__ lamb, unsigned short* __restrict__ vb) {
    size_t e = (size_t)blockIdx.x * 256 + threadIdx.x;
    float lam = lamb[0];
    int t = (int)(e >> 11), c = (int)(e & 2047);
    int h = c >> 7, hd = c & 127;
    float val = (1.f - lam) * v[e] + lam * vi[e];
    vb[((size_t)h * T_LEN + t) * HD_DIM + hd] = f32_to_bf16(val);
}

// ---------- generic bf16-WMMA GEMM:  C[M][N] = A[M][K] * W[N][K]^T ----------
// EPI: 0 = plain store, 1 = relu^2, 2 = add aux (residual)
template<int EPI>
__global__ __launch_bounds__(256) void gemm_bf16_kernel(
    const float* __restrict__ A, const float* __restrict__ W,
    float* __restrict__ C, const float* __restrict__ aux,
    int M, int N, int K) {
    constexpr int BM = 128, BN = 128, BK = 32;
    constexpr int LDA = 40, LDB = 40;           // padded strides (banks)
    __shared__ unsigned short As[BM * LDA];     // [m][k]
    __shared__ unsigned short Bs[BN * LDB];     // [n][k]
    int tid = threadIdx.x, lane = tid & 31, wid = tid >> 5;
    int bm = blockIdx.y * BM, bn = blockIdx.x * BN;
    int mBase = (wid >> 1) * 32;                // 4 waves over M
    int nBase = (wid & 1) * 64;                 // 2 waves over N

    v8f acc[2][4];
    #pragma unroll
    for (int mi = 0; mi < 2; mi++)
        #pragma unroll
        for (int ni = 0; ni < 4; ni++)
            #pragma unroll
            for (int r = 0; r < 8; r++) acc[mi][ni][r] = 0.f;

    int ldr  = tid >> 3;          // 0..31 -> used as row/8-group below
    int lr   = tid >> 3;          // row for vectorized loads (e>>3 with i offset)
    (void)ldr; (void)lr;

    for (int k0 = 0; k0 < K; k0 += BK) {
        __syncthreads();
        // vectorized stage: 1024 float4 per matrix, 4 per thread
        #pragma unroll
        for (int i = 0; i < 4; i++) {
            int e  = i * 256 + tid;
            int r  = e >> 3, c4 = (e & 7) * 4;
            float4 fa = *(const float4*)(A + (size_t)(bm + r) * K + k0 + c4);
            float4 fw = *(const float4*)(W + (size_t)(bn + r) * K + k0 + c4);
            uint2 pa; pa.x = pack2_bf16(fa.x, fa.y); pa.y = pack2_bf16(fa.z, fa.w);
            uint2 pw; pw.x = pack2_bf16(fw.x, fw.y); pw.y = pack2_bf16(fw.z, fw.w);
            *(uint2*)&As[r * LDA + c4] = pa;
            *(uint2*)&Bs[r * LDB + c4] = pw;
        }
        // prefetch next K tile into cache while WMMAs run
        if (k0 + BK < K) {
            int r = tid >> 3, c4 = (tid & 7) * 4;
            __builtin_prefetch(A + (size_t)(bm + r) * K + k0 + BK + c4, 0, 3);
            __builtin_prefetch(W + (size_t)(bn + r) * K + k0 + BK + c4, 0, 3);
        }
        __syncthreads();

        int alo = (lane & 16) ? 8 : 0;          // A frag: K chunks per ISA layout
        int blo = (lane & 16) ? 8 : 0;          // B frag: K chunks per ISA layout
        v16bf afrag[2], bfrag[4];
        #pragma unroll
        for (int mi = 0; mi < 2; mi++) {
            const unsigned short* p = &As[(mBase + mi * 16 + (lane & 15)) * LDA];
            union { v16bf v; v8bf hlf[2]; } u;
            u.hlf[0] = *(const v8bf*)(p + alo);
            u.hlf[1] = *(const v8bf*)(p + alo + 16);
            afrag[mi] = u.v;
        }
        #pragma unroll
        for (int ni = 0; ni < 4; ni++) {
            const unsigned short* p = &Bs[(nBase + ni * 16 + (lane & 15)) * LDB];
            union { v16bf v; v8bf hlf[2]; } u;
            u.hlf[0] = *(const v8bf*)(p + blo);
            u.hlf[1] = *(const v8bf*)(p + blo + 16);
            bfrag[ni] = u.v;
        }
        #pragma unroll
        for (int mi = 0; mi < 2; mi++)
            #pragma unroll
            for (int ni = 0; ni < 4; ni++)
                acc[mi][ni] = __builtin_amdgcn_wmma_f32_16x16x32_bf16(
                    false, afrag[mi], false, bfrag[ni], (short)0, acc[mi][ni], false, false);
    }

    int rowsel = (lane & 16) ? 8 : 0;           // C/D layout rows
    #pragma unroll
    for (int mi = 0; mi < 2; mi++)
        #pragma unroll
        for (int ni = 0; ni < 4; ni++)
            #pragma unroll
            for (int r = 0; r < 8; r++) {
                int m = bm + mBase + mi * 16 + rowsel + r;
                int n = bn + nBase + ni * 16 + (lane & 15);
                size_t idx = (size_t)m * N + n;
                float v = acc[mi][ni][r];
                if (EPI == 1) { v = v > 0.f ? v * v : 0.f; }
                if (EPI == 2) { v += aux[idx]; }
                C[idx] = v;
            }
}

// ---------- flash attention: one head x 128 query rows per block ----------
__global__ __launch_bounds__(256) void attn_kernel(
    const unsigned short* __restrict__ qb, const unsigned short* __restrict__ kb,
    const unsigned short* __restrict__ vb, float* __restrict__ y) {
    constexpr int LKS = 136, LVT = 40, LPS = 40;
    __shared__ unsigned short Ks[32 * LKS];     // [key][hd]
    __shared__ unsigned short Vt[128 * LVT];    // [hd][key]  (transposed for B frags)
    __shared__ unsigned short Ps[8][16 * LPS];  // per-wave P tile [row][key]
    int tid = threadIdx.x, lane = tid & 31, w = tid >> 5;
    int h = blockIdx.y;
    int q0 = blockIdx.x * 128 + w * 16;

    // persistent Q fragments (16 rows x 128 hd -> 4 K-chunks of 32)
    v16bf qf[4];
    {
        int alo = (lane & 16) ? 8 : 0;
        const unsigned short* p = qb + ((size_t)h * T_LEN + (q0 + (lane & 15))) * HD_DIM;
        #pragma unroll
        for (int c = 0; c < 4; c++) {
            union { v16bf v; v8bf hlf[2]; } u;
            u.hlf[0] = *(const v8bf*)(p + c * 32 + alo);
            u.hlf[1] = *(const v8bf*)(p + c * 32 + alo + 16);
            qf[c] = u.v;
        }
    }

    float mrow[8], lrow[8];
    v8f acc[8];
    #pragma unroll
    for (int r = 0; r < 8; r++) { mrow[r] = -1e30f; lrow[r] = 0.f; }
    #pragma unroll
    for (int f = 0; f < 8; f++)
        #pragma unroll
        for (int r = 0; r < 8; r++) acc[f][r] = 0.f;

    const float sc = 0.08838834764831845f;      // 1/sqrt(128)
    int kend = (blockIdx.x + 1) * 128;
    for (int kb0 = 0; kb0 < kend; kb0 += 32) {
        __syncthreads();
        #pragma unroll
        for (int i = 0; i < 4; i++) {
            int e = i * 256 + tid;
            int key = e >> 5, hd0 = (e & 31) * 4;
            size_t g = ((size_t)h * T_LEN + kb0 + key) * HD_DIM + hd0;
            uint2 kk = *(const uint2*)(kb + g);
            uint2 vv = *(const uint2*)(vb + g);
            *(uint2*)&Ks[key * LKS + hd0] = kk;
            Vt[(hd0 + 0) * LVT + key] = (unsigned short)(vv.x & 0xFFFFu);
            Vt[(hd0 + 1) * LVT + key] = (unsigned short)(vv.x >> 16);
            Vt[(hd0 + 2) * LVT + key] = (unsigned short)(vv.y & 0xFFFFu);
            Vt[(hd0 + 3) * LVT + key] = (unsigned short)(vv.y >> 16);
        }
        __syncthreads();
        if (kb0 <= q0 + 15) {                   // wave-uniform causal skip
            // scores: two 16x16 tiles (keys kb0.. and kb0+16..)
            v8f s01[2];
            #pragma unroll
            for (int nt = 0; nt < 2; nt++) {
                v8f s;
                #pragma unroll
                for (int r = 0; r < 8; r++) s[r] = 0.f;
                int blo = (lane & 16) ? 8 : 0;
                const unsigned short* prow = &Ks[(nt * 16 + (lane & 15)) * LKS];
                #pragma unroll
                for (int c = 0; c < 4; c++) {
                    const unsigned short* p = prow + c * 32 + blo;
                    union { v16bf v; v8bf hlf[2]; } u;
                    u.hlf[0] = *(const v8bf*)p;
                    u.hlf[1] = *(const v8bf*)(p + 16);
                    s = __builtin_amdgcn_wmma_f32_16x16x32_bf16(
                        false, qf[c], false, u.v, (short)0, s, false, false);
                }
                s01[nt] = s;
            }
            int rowoff = (lane & 16) ? 8 : 0;
            float alpha[8];
            #pragma unroll
            for (int r = 0; r < 8; r++) {
                int qi = q0 + rowoff + r;
                float v0 = s01[0][r] * sc; if (kb0 +      (lane & 15) > qi) v0 = -1e30f;
                float v1 = s01[1][r] * sc; if (kb0 + 16 + (lane & 15) > qi) v1 = -1e30f;
                float rmax = fmaxf(v0, v1);
                #pragma unroll
                for (int mm = 8; mm >= 1; mm >>= 1) rmax = fmaxf(rmax, __shfl_xor(rmax, mm, 32));
                float mnew = fmaxf(mrow[r], rmax);
                float p0 = __expf(v0 - mnew), p1 = __expf(v1 - mnew);
                float a0 = __expf(mrow[r] - mnew);
                float rs = p0 + p1;
                #pragma unroll
                for (int mm = 8; mm >= 1; mm >>= 1) rs += __shfl_xor(rs, mm, 32);
                lrow[r] = lrow[r] * a0 + rs;
                mrow[r] = mnew; alpha[r] = a0;
                Ps[w][(rowoff + r) * LPS +      (lane & 15)] = f32_to_bf16(p0);
                Ps[w][(rowoff + r) * LPS + 16 + (lane & 15)] = f32_to_bf16(p1);
            }
            #pragma unroll
            for (int f = 0; f < 8; f++)
                #pragma unroll
                for (int r = 0; r < 8; r++) acc[f][r] *= alpha[r];
            // reload P as A-layout fragment (D-layout -> A-layout via LDS)
            union { v16bf v; v8bf hlf[2]; } up;
            {
                int alo = (lane & 16) ? 8 : 0;
                const unsigned short* p = &Ps[w][(lane & 15) * LPS + alo];
                up.hlf[0] = *(const v8bf*)p;
                up.hlf[1] = *(const v8bf*)(p + 16);
            }
            #pragma unroll
            for (int f = 0; f < 8; f++) {
                int klo = (lane & 16) ? 8 : 0;
                const unsigned short* p = &Vt[(f * 16 + (lane & 15)) * LVT + klo];
                union { v16bf v; v8bf hlf[2]; } uv;
                uv.hlf[0] = *(const v8bf*)p;
                uv.hlf[1] = *(const v8bf*)(p + 16);
                acc[f] = __builtin_amdgcn_wmma_f32_16x16x32_bf16(
                    false, up.v, false, uv.v, (short)0, acc[f], false, false);
            }
        }
    }
    int rowoff = (lane & 16) ? 8 : 0;
    #pragma unroll
    for (int r = 0; r < 8; r++) {
        float inv = 1.f / lrow[r];
        int qr = q0 + rowoff + r;
        #pragma unroll
        for (int f = 0; f < 8; f++)
            y[(size_t)qr * D_DIM + h * HD_DIM + f * 16 + (lane & 15)] = acc[f][r] * inv;
    }
}

// ---------- host side ----------
extern "C" void kernel_launch(void* const* d_in, const int* in_sizes, int n_in,
                              void* d_out, int out_size, void* d_ws, size_t ws_size,
                              hipStream_t stream) {
    (void)in_sizes; (void)n_in; (void)out_size; (void)ws_size;
    const float* x       = (const float*)d_in[0];
    const float* vi      = (const float*)d_in[1];
    const float* x0      = (const float*)d_in[2];
    const float* wq      = (const float*)d_in[3];
    const float* wk      = (const float*)d_in[4];
    const float* wv      = (const float*)d_in[5];
    const float* wo      = (const float*)d_in[6];
    const float* lamb    = (const float*)d_in[7];
    const float* lambdas = (const float*)d_in[8];
    const float* w_fc    = (const float*)d_in[9];
    const float* w_proj  = (const float*)d_in[10];
    float* out = (float*)d_out;

    const size_t MB = 1024u * 1024u;
    char* ws = (char*)d_ws;
    float* xn = (float*)(ws + 0 * MB);
    float* q  = (float*)(ws + 16 * MB);
    float* k  = (float*)(ws + 32 * MB);
    float* v  = (float*)(ws + 48 * MB);
    float* y  = (float*)(ws + 64 * MB);
    float* a  = (float*)(ws + 80 * MB);
    float* an = (float*)(ws + 96 * MB);
    float* hb = (float*)(ws + 112 * MB);                // [T][4D], 64 MB
    unsigned short* qb = (unsigned short*)(ws + 176 * MB);
    unsigned short* kb = (unsigned short*)(ws + 184 * MB);
    unsigned short* vb = (unsigned short*)(ws + 192 * MB);

    dim3 blk(256);
    // 1. xb = l0*x + l1*x0 ; xn = rms_norm(xb)
    blend_rmsnorm_kernel<<<T_LEN, blk, 0, stream>>>(x, x0, lambdas, xn);
    // 2. q,k,v projections
    dim3 gD(D_DIM / 128, T_LEN / 128);
    gemm_bf16_kernel<0><<<gD, blk, 0, stream>>>(xn, wq, q, nullptr, T_LEN, D_DIM, D_DIM);
    gemm_bf16_kernel<0><<<gD, blk, 0, stream>>>(xn, wk, k, nullptr, T_LEN, D_DIM, D_DIM);
    gemm_bf16_kernel<0><<<gD, blk, 0, stream>>>(xn, wv, v, nullptr, T_LEN, D_DIM, D_DIM);
    // 3. per-head rms_norm + rope on q,k ; value blend -> head-major bf16
    rope_norm_kernel<<<T_LEN, blk, 0, stream>>>(q, qb);
    rope_norm_kernel<<<T_LEN, blk, 0, stream>>>(k, kb);
    vblend_kernel<<<(T_LEN * D_DIM) / 256, blk, 0, stream>>>(v, vi, lamb, vb);
    // 4. causal flash attention
    attn_kernel<<<dim3(T_LEN / 128, H_NUM), blk, 0, stream>>>(qb, kb, vb, y);
    // 5. output projection (replaces x, no residual)
    gemm_bf16_kernel<0><<<gD, blk, 0, stream>>>(y, wo, a, nullptr, T_LEN, D_DIM, D_DIM);
    // 6. MLP: rms_norm -> fc (relu^2 fused) -> proj (+a fused), writes d_out
    rmsnorm_kernel<<<T_LEN, blk, 0, stream>>>(a, an);
    gemm_bf16_kernel<1><<<dim3(4 * D_DIM / 128, T_LEN / 128), blk, 0, stream>>>(
        an, w_fc, hb, nullptr, T_LEN, 4 * D_DIM, D_DIM);
    gemm_bf16_kernel<2><<<gD, blk, 0, stream>>>(
        hb, w_proj, out, a, T_LEN, D_DIM, 4 * D_DIM);
}
